// DescripNet_42743514530238
// MI455X (gfx1250) — matrix operand order repaired and checked by reading, not compiled, not run on hardware
//
#include <hip/hip_runtime.h>

// DGCNN-style DescripNet for MI455X (gfx1250, wave32, WMMA).
//  - All big matmuls (kNN distances h@h^T, t=h@theta, u=h@(phi-theta), feat GEMM)
//    run on v_wmma_f32_16x16x32_bf16 (bf16 in, f32 accumulate).
//  - Distance GEMM reads an explicitly transposed feature copy (hbT) so B
//    fragments are 2-byte loads, 32B contiguous across 16 lanes per k row.
//  - GEMM waves compute 16x64 strips: 1 A-frag feeds 4 WMMAs per K-step.
//  - EdgeConv rewritten as gather-max over precomputed t-table (L2-resident).
//  - kNN fused: 1 wave per 16-point stripe, register-resident top-20.

#define N_PTS 8192
#define KNN   20
#define HB_STRIDE 512   // fixed row stride for all NxC feature buffers

typedef __attribute__((ext_vector_type(16))) __bf16 v16bf;
typedef __attribute__((ext_vector_type(8)))  __bf16 v8bf;
typedef __attribute__((ext_vector_type(8)))  float  v8f;

// ---- WMMA fragment loaders (layouts per CDNA5 ISA 7.12.2) ------------------

// A (16x32 bf16, MxK): lane<16 -> K octets {0..7,16..23}; lane>=16 -> {8..15,24..31}
__device__ inline v16bf load_a_frag(const __bf16* A, int lda, int row0, int k0, int lane) {
  const int m  = lane & 15;
  const int kh = (lane < 16) ? 0 : 8;
  const __bf16* p = A + (size_t)(row0 + m) * lda + k0 + kh;
  v8bf lo = *(const v8bf*)(p);
  v8bf hi = *(const v8bf*)(p + 16);
  v16bf r;
#pragma unroll
  for (int i = 0; i < 8; ++i) { r[i] = lo[i]; r[8 + i] = hi[i]; }
  return r;
}

// B (32x16 bf16, KxN) row-major: lane = N column; lane<16 -> K {0..7,16..23}; lane>=16 -> {8..15,24..31}
__device__ inline v16bf load_b_frag(const __bf16* B, int ldb, int k0, int n0, int lane) {
  const int kh = (lane < 16) ? 0 : 8;
  const __bf16* p = B + (size_t)(k0 + kh) * ldb + n0 + (lane & 15);
  v16bf r;
#pragma unroll
  for (int i = 0; i < 8; ++i) { r[i] = p[(size_t)i * ldb]; r[8 + i] = p[(size_t)(16 + i) * ldb]; }
  return r;
}

// ---- Setup / small kernels -------------------------------------------------

__global__ void cast_x_kernel(const float* __restrict__ x, __bf16* __restrict__ hb) {
  int id = blockIdx.x * blockDim.x + threadIdx.x;
  if (id >= N_PTS * 32) return;
  int i = id >> 5, c = id & 31;
  float v = (c < 3) ? x[i * 3 + c] : 0.0f;  // pad K to 32 with zeros
  hb[(size_t)i * HB_STRIDE + c] = (__bf16)v;
}

// LDS-tiled transpose: hbT[c][i] = hb[i][c], c < kp. Coalesced both sides.
__global__ __launch_bounds__(256) void transpose_kernel(const __bf16* __restrict__ hb,
                                                        __bf16* __restrict__ hbT) {
  __shared__ __bf16 t[32][33];
  const int i0 = blockIdx.x * 32;
  const int c0 = blockIdx.y * 32;
  const int tx = threadIdx.x & 31;
  const int ty = threadIdx.x >> 5;  // 0..7
#pragma unroll
  for (int r = 0; r < 32; r += 8)
    t[r + ty][tx] = hb[(size_t)(i0 + r + ty) * HB_STRIDE + c0 + tx];
  __syncthreads();
#pragma unroll
  for (int r = 0; r < 32; r += 8)
    hbT[(size_t)(c0 + r + ty) * N_PTS + i0 + tx] = t[tx][r + ty];
}

__global__ void sqnorm_kernel(const __bf16* __restrict__ hb, float* __restrict__ sq, int kp) {
  int i = blockIdx.x * blockDim.x + threadIdx.x;
  if (i >= N_PTS) return;
  const __bf16* p = hb + (size_t)i * HB_STRIDE;
  float s = 0.0f;
  for (int c = 0; c < kp; ++c) { float v = (float)p[c]; s += v * v; }
  sq[i] = s;
}

__global__ void prep_layer_kernel(const float* __restrict__ tw, const float* __restrict__ tb,
                                  const float* __restrict__ pw, const float* __restrict__ pb,
                                  const float* __restrict__ bng, const float* __restrict__ bnb,
                                  const float* __restrict__ bnm, const float* __restrict__ bnv,
                                  __bf16* __restrict__ wt, __bf16* __restrict__ wc,
                                  float* __restrict__ biasc, float* __restrict__ scale,
                                  float* __restrict__ shift, int din, int dout, int kp) {
  int id = blockIdx.x * blockDim.x + threadIdx.x;
  int total = kp * dout;
  if (id < total) {
    int k = id / dout, c = id % dout;
    float t = (k < din) ? tw[k * dout + c] : 0.0f;
    float p = (k < din) ? pw[k * dout + c] : 0.0f;
    wt[id] = (__bf16)t;
    wc[id] = (__bf16)(p - t);
  }
  if (id < dout) {
    float sc = bng[id] * rsqrtf(bnv[id] + 1e-5f);
    scale[id] = sc;
    shift[id] = bnb[id] - bnm[id] * sc;
    biasc[id] = tb[id] + pb[id];
  }
}

__global__ void cast_w_kernel(const float* __restrict__ w, __bf16* __restrict__ wb, int n) {
  int id = blockIdx.x * blockDim.x + threadIdx.x;
  if (id < n) wb[id] = (__bf16)w[id];
}

// ---- Fused kNN: 1 wave per 16-row stripe, register top-20 ------------------
// Distances: D[m][n] = sq[m] + sq[n] - 2 * (h h^T)[m][n]; B frags come from hbT.

template <int KP>
__global__ __launch_bounds__(32) void knn_kernel(const __bf16* __restrict__ hb,
                                                 const __bf16* __restrict__ hbT,
                                                 const float* __restrict__ sq,
                                                 int* __restrict__ idx) {
  __shared__ float tile[16][17];
  const int lane = threadIdx.x;
  const int row0 = blockIdx.x * 16;
  constexpr int NK = KP / 32;

  v16bf af[NK];
#pragma unroll
  for (int kc = 0; kc < NK; ++kc) af[kc] = load_a_frag(hb, HB_STRIDE, row0, kc * 32, lane);

  const int nl = lane & 15;
  const int mt = (lane < 16) ? 0 : 8;
  float sqa[8];
#pragma unroll
  for (int v = 0; v < 8; ++v) sqa[v] = sq[row0 + mt + v];

  float bd[KNN]; int bi[KNN];           // sorted descending: bd[0] = current worst
#pragma unroll
  for (int t = 0; t < KNN; ++t) { bd[t] = 3.4e38f; bi[t] = 0; }

  for (int n0 = 0; n0 < N_PTS; n0 += 16) {
    v8f c = {};
#pragma unroll
    for (int kc = 0; kc < NK; ++kc) {
      // B[k][n] = hbT[k][n0+n]: 32B contiguous across 16 lanes per k row
      v16bf b = load_b_frag(hbT, N_PTS, kc * 32, n0, lane);
      c = __builtin_amdgcn_wmma_f32_16x16x32_bf16(false, af[kc], false, b,
                                                  (short)0, c, false, false);
    }
    float sqn = sq[n0 + nl];
#pragma unroll
    for (int v = 0; v < 8; ++v)
      tile[mt + v][nl] = sqa[v] + sqn - 2.0f * c[v];
    __syncthreads();

    if (lane < 16) {
      for (int n = 0; n < 16; ++n) {
        float d = tile[lane][n];
        if (d < bd[0]) {
          float cv = d; int ci = n0 + n; bool placed = false;
#pragma unroll
          for (int s = 0; s < KNN; ++s) {
            if (!placed) {
              if (s + 1 < KNN && cv < bd[s + 1]) { bd[s] = bd[s + 1]; bi[s] = bi[s + 1]; }
              else { bd[s] = cv; bi[s] = ci; placed = true; }
            }
          }
        }
      }
    }
    __syncthreads();
  }
  if (lane < 16) {
#pragma unroll
    for (int t = 0; t < KNN; ++t) idx[(size_t)(row0 + lane) * KNN + t] = bi[t];
  }
}

// ---- GEMM: C[N x ncols] = A[N x kp](bf16) * B[kp x ncols](bf16) ------------
// One wave computes a 16x64 strip: A-frag reused across 4 WMMAs per K-step.

__global__ __launch_bounds__(256) void gemm_bf16_kernel(const __bf16* __restrict__ A,
                                                        const __bf16* __restrict__ B,
                                                        float* __restrict__ C,
                                                        int ncols, int kp,
                                                        const float* __restrict__ bias,
                                                        int relu) {
  const int gt   = blockIdx.x * blockDim.x + threadIdx.x;
  const int wave = gt >> 5;
  const int lane = gt & 31;
  const int spr  = ncols >> 6;              // 64-wide strips per row band
  const int row0 = (wave / spr) << 4;
  const int n0   = (wave % spr) << 6;
  if (row0 >= N_PTS) return;

  v8f c0 = {}, c1 = {}, c2 = {}, c3 = {};
  for (int k0 = 0; k0 < kp; k0 += 32) {
    v16bf a  = load_a_frag(A, HB_STRIDE, row0, k0, lane);
    v16bf b0 = load_b_frag(B, ncols, k0, n0 +  0, lane);
    v16bf b1 = load_b_frag(B, ncols, k0, n0 + 16, lane);
    v16bf b2 = load_b_frag(B, ncols, k0, n0 + 32, lane);
    v16bf b3 = load_b_frag(B, ncols, k0, n0 + 48, lane);
    c0 = __builtin_amdgcn_wmma_f32_16x16x32_bf16(false, a, false, b0, (short)0, c0, false, false);
    c1 = __builtin_amdgcn_wmma_f32_16x16x32_bf16(false, a, false, b1, (short)0, c1, false, false);
    c2 = __builtin_amdgcn_wmma_f32_16x16x32_bf16(false, a, false, b2, (short)0, c2, false, false);
    c3 = __builtin_amdgcn_wmma_f32_16x16x32_bf16(false, a, false, b3, (short)0, c3, false, false);
  }
  const int nl = lane & 15;
  const int mt = (lane < 16) ? 0 : 8;
  v8f acc[4] = {c0, c1, c2, c3};
#pragma unroll
  for (int j = 0; j < 4; ++j) {
    const int n = n0 + 16 * j + nl;
    const float bv = bias ? bias[n] : 0.0f;
#pragma unroll
    for (int v = 0; v < 8; ++v) {
      float x = acc[j][v] + bv;
      if (relu) x = fmaxf(x, 0.0f);
      C[(size_t)(row0 + mt + v) * ncols + n] = x;
    }
  }
}

// ---- EdgeConv epilogue: gather-max + BN + leaky, write next-layer bf16 -----

__global__ void edge_max_kernel(const float* __restrict__ t, const float* __restrict__ u,
                                const int* __restrict__ idx,
                                const float* __restrict__ scale, const float* __restrict__ shift,
                                __bf16* __restrict__ hb, int dout) {
  int id = blockIdx.x * blockDim.x + threadIdx.x;
  int total = N_PTS * dout;
  if (id >= total) return;
  int i = id / dout, c = id % dout;
  const int* ip = idx + (size_t)i * KNN;
  float best = -3.4e38f;
#pragma unroll 4
  for (int j = 0; j < KNN; ++j)
    best = fmaxf(best, t[(size_t)ip[j] * dout + c]);
  float m = (best + u[id]) * scale[c] + shift[c];  // BN commutes with max (scale>0)
  m = (m > 0.0f) ? m : 0.2f * m;                   // leaky relu
  hb[(size_t)i * HB_STRIDE + c] = (__bf16)m;
}

// ---- Head: gate, softmax over points, gated sum ----------------------------

__global__ void gate_kernel(const __bf16* __restrict__ hb, const float* __restrict__ gw,
                            const float* __restrict__ gb, float* __restrict__ g) {
  int i = blockIdx.x * blockDim.x + threadIdx.x;
  if (i >= N_PTS) return;
  const __bf16* p = hb + (size_t)i * HB_STRIDE;
  float acc = gb[0];
  for (int c = 0; c < 512; ++c) acc += (float)p[c] * gw[c];
  g[i] = fmaxf(acc, 0.0f);
}

__global__ __launch_bounds__(1024) void softmax_kernel(const float* __restrict__ g,
                                                       float* __restrict__ e,
                                                       float* __restrict__ red) {
  __shared__ float s[1024];
  int t = threadIdx.x;
  float m = -3.4e38f;
  for (int i = t; i < N_PTS; i += 1024) m = fmaxf(m, g[i]);
  s[t] = m; __syncthreads();
  for (int o = 512; o > 0; o >>= 1) { if (t < o) s[t] = fmaxf(s[t], s[t + o]); __syncthreads(); }
  float mm = s[0]; __syncthreads();
  float sum = 0.0f;
  for (int i = t; i < N_PTS; i += 1024) { float ex = __expf(g[i] - mm); e[i] = ex; sum += ex; }
  s[t] = sum; __syncthreads();
  for (int o = 512; o > 0; o >>= 1) { if (t < o) s[t] += s[t + o]; __syncthreads(); }
  if (t == 0) red[0] = s[0];
}

__global__ __launch_bounds__(256) void out_kernel(const float* __restrict__ feat,
                                                  const float* __restrict__ e,
                                                  const float* __restrict__ red,
                                                  float* __restrict__ out) {
  __shared__ float s[256];
  int c = blockIdx.x, t = threadIdx.x;
  float acc = 0.0f;
  for (int i = t; i < N_PTS; i += 256) acc += e[i] * feat[(size_t)i * 512 + c];
  s[t] = acc; __syncthreads();
  for (int o = 128; o > 0; o >>= 1) { if (t < o) s[t] += s[t + o]; __syncthreads(); }
  if (t == 0) out[c] = s[0] / red[0];
}

// ---- Host orchestration ----------------------------------------------------

extern "C" void kernel_launch(void* const* d_in, const int* in_sizes, int n_in,
                              void* d_out, int out_size, void* d_ws, size_t ws_size,
                              hipStream_t stream) {
  // Input flattening: {'x', 'params'}; params pytree-sorted:
  //   convs[0..4] each: bn_b, bn_g, bn_mean, bn_var, phi_b, phi_w, theta_b, theta_w
  //   then feat_b, feat_w, gate_b, gate_w
  const float* x = (const float*)d_in[0];
  const float* feat_b = (const float*)d_in[41];
  const float* feat_w = (const float*)d_in[42];
  const float* gate_b = (const float*)d_in[43];
  const float* gate_w = (const float*)d_in[44];

  // Workspace carve-up (~52 MB)
  char* wp = (char*)d_ws;
  auto alloc = [&](size_t bytes) { void* r = (void*)wp; wp += (bytes + 255) & ~(size_t)255; return r; };
  __bf16* hb    = (__bf16*)alloc((size_t)N_PTS * 512 * 2);
  __bf16* hbT   = (__bf16*)alloc((size_t)512 * N_PTS * 2);
  float*  tbuf  = (float*) alloc((size_t)N_PTS * 512 * 4);
  float*  ubuf  = (float*) alloc((size_t)N_PTS * 512 * 4);
  float*  sq    = (float*) alloc((size_t)N_PTS * 4);
  int*    idx   = (int*)   alloc((size_t)N_PTS * KNN * 4);
  __bf16* wt    = (__bf16*)alloc((size_t)512 * 512 * 2);
  __bf16* wc    = (__bf16*)alloc((size_t)512 * 512 * 2);
  float*  biasc = (float*) alloc(512 * 4);
  float*  scale = (float*) alloc(512 * 4);
  float*  shift = (float*) alloc(512 * 4);
  float*  gbuf  = (float*) alloc((size_t)N_PTS * 4);
  float*  ebuf  = (float*) alloc((size_t)N_PTS * 4);
  float*  red   = (float*) alloc(16);

  auto launch_gemm = [&](const __bf16* A, const __bf16* B, float* C, int ncols, int kp,
                         const float* bias, int relu) {
    int waves = (N_PTS / 16) * (ncols / 64);
    int threads = waves * 32;
    gemm_bf16_kernel<<<(threads + 255) / 256, 256, 0, stream>>>(A, B, C, ncols, kp, bias, relu);
  };

  // Embed input (pad 3 -> 32 with zeros)
  cast_x_kernel<<<(N_PTS * 32 + 255) / 256, 256, 0, stream>>>(x, hb);

  const int din[5]  = {3, 64, 64, 128, 256};
  const int dout[5] = {64, 64, 128, 256, 512};
  const int kp[5]   = {32, 64, 64, 128, 256};

  for (int L = 0; L < 5; ++L) {
    const int base = 1 + 8 * L;
    const float* bn_b  = (const float*)d_in[base + 0];
    const float* bn_g  = (const float*)d_in[base + 1];
    const float* bn_m  = (const float*)d_in[base + 2];
    const float* bn_v  = (const float*)d_in[base + 3];
    const float* phi_b = (const float*)d_in[base + 4];
    const float* phi_w = (const float*)d_in[base + 5];
    const float* th_b  = (const float*)d_in[base + 6];
    const float* th_w  = (const float*)d_in[base + 7];

    // kNN on current features (transposed copy feeds distance-GEMM B frags)
    sqnorm_kernel<<<(N_PTS + 255) / 256, 256, 0, stream>>>(hb, sq, kp[L]);
    {
      dim3 tg(N_PTS / 32, kp[L] / 32);
      transpose_kernel<<<tg, 256, 0, stream>>>(hb, hbT);
    }
    switch (kp[L]) {
      case 32:  knn_kernel<32> <<<N_PTS / 16, 32, 0, stream>>>(hb, hbT, sq, idx); break;
      case 64:  knn_kernel<64> <<<N_PTS / 16, 32, 0, stream>>>(hb, hbT, sq, idx); break;
      case 128: knn_kernel<128><<<N_PTS / 16, 32, 0, stream>>>(hb, hbT, sq, idx); break;
      case 256: knn_kernel<256><<<N_PTS / 16, 32, 0, stream>>>(hb, hbT, sq, idx); break;
    }

    // Pack weights (bf16, zero-padded K) + fold biases & BN constants
    {
      int total = kp[L] * dout[L];
      prep_layer_kernel<<<(total + 255) / 256, 256, 0, stream>>>(
          th_w, th_b, phi_w, phi_b, bn_g, bn_b, bn_m, bn_v,
          wt, wc, biasc, scale, shift, din[L], dout[L], kp[L]);
    }

    // t = h @ theta ; u = h @ (phi - theta) + (theta_b + phi_b)
    launch_gemm(hb, wt, tbuf, dout[L], kp[L], nullptr, 0);
    launch_gemm(hb, wc, ubuf, dout[L], kp[L], biasc, 0);

    // h_next = leaky( BN( max_j t[idx] + u ) )  -> overwrites hb
    {
      int total = N_PTS * dout[L];
      edge_max_kernel<<<(total + 255) / 256, 256, 0, stream>>>(
          tbuf, ubuf, idx, scale, shift, hb, dout[L]);
    }
  }

  // Head: gate = relu(h@gw+gb); softmax over points; feat = relu(h@fw+fb); gated sum
  gate_kernel<<<(N_PTS + 255) / 256, 256, 0, stream>>>(hb, gate_w, gate_b, gbuf);
  softmax_kernel<<<1, 1024, 0, stream>>>(gbuf, ebuf, red);
  cast_w_kernel<<<(512 * 512 + 255) / 256, 256, 0, stream>>>(feat_w, wt, 512 * 512);
  launch_gemm(hb, wt, tbuf, 512, 512, feat_b, 1);
  out_kernel<<<512, 256, 0, stream>>>(tbuf, ebuf, red, (float*)d_out);
}
